// FEM0227_55439437857355
// MI455X (gfx1250) — matrix operand order repaired and checked by reading, not compiled, not run on hardware
//
#include <hip/hip_runtime.h>

typedef __attribute__((ext_vector_type(16))) _Float16 v16h;
typedef __attribute__((ext_vector_type(8)))  float    v8f;
typedef __attribute__((ext_vector_type(4)))  unsigned int v4u;
typedef __attribute__((ext_vector_type(8)))  int      v8i;
typedef __attribute__((ext_vector_type(4)))  int      v4i;
typedef _Float16 h16;

#define NPIX 4096
#define CIN  256
#define CI   128
#define BEPS 1e-5f

// K-index inside a 16x32 f16 A-fragment for element e, lane-half kh
// (contiguous order: element e of lane-half kh holds K = kh*16 + e, matching the B fills)
__device__ __forceinline__ int a_kidx(int e, int kh) {
  return kh * 16 + e;
}

__device__ __forceinline__ v8f wmma16(v16h a, v16h b, v8f c) {
  return __builtin_amdgcn_wmma_f32_16x16x32_f16(false, a, false, b, (short)0, c, false, false);
}

// ---------------------------------------------------------------------------
// Tensor Data Mover: DMA a 64-row x 128-half tile (row stride 128 halves in
// global) into LDS with 16B padding after every 256B row -> LDS pitch = 136
// halves. D# built per CDNA5 ISA 8.3/8.4 bitfields; 2D mode (groups 2/3 zero).
// clang-23 toolchain: 6-arg builtin (g0, g1, g2, g3, extra, cpol).
// ---------------------------------------------------------------------------
__device__ __forceinline__ void tdm_load_tile_64x128(unsigned lds_off, const h16* gptr) {
  unsigned long long ga = (unsigned long long)(uintptr_t)gptr;
  v4u g0;
  g0[0] = 1u;                                   // count=1, user descriptor
  g0[1] = lds_off;                              // lds_addr (bytes)
  g0[2] = (unsigned)(ga & 0xffffffffu);         // global_addr[31:0]
  g0[3] = (unsigned)((ga >> 32) & 0x01ffffffu)  // global_addr[56:32]
        | (2u << 30);                           // type = 2 ("image")
  v8i g1;
  g1[0] = (int)((1u << 16)                      // data_size = 2 bytes
              | (1u << 20)                      // pad_enable
              | (5u << 22)                      // pad_interval: every 64 DWORDs
              | (3u << 25));                    // pad_amount: 4 DWORDs (16B)
  g1[1] = (int)(128u << 16);                    // tensor_dim0 = 128 elems
  g1[2] = (int)(64u << 16);                     // tensor_dim1 = 64 rows
  g1[3] = (int)(128u << 16);                    // tile_dim0 = 128
  g1[4] = 64;                                   // tile_dim1 = 64, tile_dim2 = 0
  g1[5] = 128;                                  // tensor_dim0_stride = 128
  g1[6] = 0;
  g1[7] = 0;
  v4i z4 = {0, 0, 0, 0};
  v8i z8 = {0, 0, 0, 0, 0, 0, 0, 0};
  __builtin_amdgcn_tensor_load_to_lds(g0, g1, z4, z4, z8, 0);
}

// ---------------------------------------------------------------------------
// 1x1-conv projection GEMM: out[o,n] = sum_c W[o,c] * X[c,n] + bias[o], opt BN.
// Block = 128 thr (4 waves), tile = 16 o x 64 n, K looped in steps of 32.
// ---------------------------------------------------------------------------
__global__ void __launch_bounds__(128)
proj_gemm(const float* __restrict__ X, const float* __restrict__ Wm,
          const float* __restrict__ bias,
          const float* __restrict__ bg, const float* __restrict__ bb,
          const float* __restrict__ bm, const float* __restrict__ bv,
          float* __restrict__ out32,      // [B][O][NPIX] f32 (or null)
          h16*   __restrict__ outC16,     // [B][O][NPIX] f16 ch-major (or null)
          h16*   __restrict__ outP16,     // [B][NPIX][CI] f16 pix-major (or null)
          int O, int colOff)
{
  __shared__ h16 xt[64][40];              // X tile transposed: [n][c]
  const int tid  = threadIdx.x;
  const int lane = tid & 31;
  const int wv   = tid >> 5;
  const int kh   = lane >> 4;
  const int l15  = lane & 15;
  const int n0   = blockIdx.x * 64;
  const int o0   = blockIdx.y * 16;
  const int b    = blockIdx.z;
  const float* Xb   = X  + (size_t)b * CIN * NPIX;
  const float* Wrow = Wm + (size_t)(o0 + l15) * CIN;

  v8f acc = {};
  for (int kk = 0; kk < CIN; kk += 32) {
    __syncthreads();
    #pragma unroll
    for (int it = 0; it < 16; ++it) {
      int idx = tid + it * 128;
      int nl = idx & 63, cl = idx >> 6;
      xt[nl][cl] = (h16)Xb[(size_t)(kk + cl) * NPIX + n0 + nl];
    }
    __syncthreads();
    v16h a, bf;
    #pragma unroll
    for (int e = 0; e < 16; ++e) a[e]  = (h16)Wrow[kk + a_kidx(e, kh)];
    #pragma unroll
    for (int e = 0; e < 16; ++e) bf[e] = xt[wv * 16 + l15][(kh << 4) + e];
    acc = wmma16(a, bf, acc);
  }

  const int rbase = kh * 8;
  #pragma unroll
  for (int r = 0; r < 8; ++r) {
    int o = o0 + rbase + r;
    int n = n0 + wv * 16 + l15;
    float x = acc[r] + bias[o];
    if (bg) {
      float inv = bg[o] * rsqrtf(bv[o] + BEPS);
      x = (x - bm[o]) * inv + bb[o];
    }
    if (out32)  out32 [(size_t)b * O * NPIX + (size_t)o * NPIX + n] = x;
    if (outC16) outC16[(size_t)b * O * NPIX + (size_t)o * NPIX + n] = (h16)x;
    if (outP16) outP16[(size_t)b * NPIX * CI + (size_t)n * CI + colOff + o] = (h16)x;
  }
}

// ---------------------------------------------------------------------------
// Fused flash attention:  E = scale * softmax(Q Kt) V + T
// Qp,Kp pixel-major f16 [B][N][128]; Vc, Tadd ch-major [B][128][N].
// Block = 64 rows (4 waves x 16), online softmax, full column sweep.
// Q tile and double-buffered K tiles are DMA'd into LDS by the Tensor Data
// Mover (wave 0 issues, TENSORcnt + block barrier synchronize), so the K
// stream for block i+1 overlaps the WMMAs of block i.
// ---------------------------------------------------------------------------
__global__ void __launch_bounds__(128)
flash_attn(const h16* __restrict__ Qp, const h16* __restrict__ Kp,
           const h16* __restrict__ Vc, const float* __restrict__ Tadd,
           const float* __restrict__ scale_p,
           float* __restrict__ Eout, h16* __restrict__ E16)
{
  __shared__ h16 Qs[64][136];
  __shared__ h16 Ks2[2][64][136];
  __shared__ h16 Ps[4][16][72];           // per-wave P tile (C-layout -> A-layout bridge)
  const int tid   = threadIdx.x;
  const int lane  = tid & 31;
  const int wv    = tid >> 5;
  const int kh    = lane >> 4;
  const int l15   = lane & 15;
  const int rbase = kh * 8;
  const int n0    = blockIdx.x * 64;
  const int b     = blockIdx.y;
  const size_t bpix = (size_t)b * NPIX * CI;
  const size_t bch  = (size_t)b * CI * NPIX;

  const unsigned qs_lds = (unsigned)(uintptr_t)&Qs[0][0];
  const unsigned ks_lds0 = (unsigned)(uintptr_t)&Ks2[0][0][0];
  const unsigned ks_lds1 = (unsigned)(uintptr_t)&Ks2[1][0][0];

  // Prologue: async-DMA Q tile and first K tile (wave 0 only; TDM ignores EXEC)
  if (wv == 0) {
    tdm_load_tile_64x128(qs_lds, Qp + bpix + (size_t)n0 * CI);
    tdm_load_tile_64x128(ks_lds0, Kp + bpix);
  }

  v8f Oacc[8] = {};
  float mrow[8], lrow[8];
  #pragma unroll
  for (int r = 0; r < 8; ++r) { mrow[r] = -3.0e38f; lrow[r] = 0.f; }

  for (int it = 0; it < NPIX / 64; ++it) {
    const int m0 = it * 64;
    const int p  = it & 1;
    if (wv == 0) __builtin_amdgcn_s_wait_tensorcnt(0);  // buffer p (and Q) landed
    __syncthreads();                                    // release compute waves;
                                                        // also: all done reading p^1
    if (wv == 0 && m0 + 64 < NPIX)
      tdm_load_tile_64x128(p ? ks_lds0 : ks_lds1,
                           Kp + bpix + (size_t)(m0 + 64) * CI);
    if (m0 + 64 < NPIX)
      __builtin_prefetch(Vc + bch + (size_t)(tid & 127) * NPIX + m0 + 64, 0, 1);

    // S = Q * K^T for this 64-column block (4 col tiles per wave)
    v8f sacc[4] = {};
    #pragma unroll
    for (int kkb = 0; kkb < 4; ++kkb) {
      v16h a;
      #pragma unroll
      for (int e = 0; e < 16; ++e)
        a[e] = Qs[wv * 16 + l15][kkb * 32 + a_kidx(e, kh)];
      #pragma unroll
      for (int jt = 0; jt < 4; ++jt) {
        v16h bf;
        #pragma unroll
        for (int e = 0; e < 16; ++e)
          bf[e] = Ks2[p][jt * 16 + l15][kkb * 32 + (kh << 4) + e];
        sacc[jt] = wmma16(a, bf, sacc[jt]);
      }
    }

    // online softmax: rows live as register slot r, columns across 16-lane groups
    #pragma unroll
    for (int r = 0; r < 8; ++r) {
      float mx = fmaxf(fmaxf(sacc[0][r], sacc[1][r]), fmaxf(sacc[2][r], sacc[3][r]));
      #pragma unroll
      for (int msk = 8; msk >= 1; msk >>= 1) mx = fmaxf(mx, __shfl_xor(mx, msk, 32));
      float mnew = fmaxf(mrow[r], mx);
      float fac  = __expf(mrow[r] - mnew);
      mrow[r] = mnew;
      float psum = 0.f;
      #pragma unroll
      for (int jt = 0; jt < 4; ++jt) {
        float ppv = __expf(sacc[jt][r] - mnew);
        sacc[jt][r] = ppv;
        psum += ppv;
      }
      #pragma unroll
      for (int msk = 8; msk >= 1; msk >>= 1) psum += __shfl_xor(psum, msk, 32);
      lrow[r] = lrow[r] * fac + psum;
      #pragma unroll
      for (int ct = 0; ct < 8; ++ct) Oacc[ct][r] *= fac;
    }

    // park P in LDS (f16) to re-read it in A-fragment layout
    #pragma unroll
    for (int jt = 0; jt < 4; ++jt)
      #pragma unroll
      for (int r = 0; r < 8; ++r)
        Ps[wv][rbase + r][jt * 16 + l15] = (h16)sacc[jt][r];
    __syncthreads();

    // O += P * V  (V B-fragments straight from global: ch-major rows are m-contiguous)
    #pragma unroll
    for (int ms = 0; ms < 2; ++ms) {
      v16h pa;
      #pragma unroll
      for (int e = 0; e < 16; ++e)
        pa[e] = Ps[wv][l15][ms * 32 + a_kidx(e, kh)];
      #pragma unroll
      for (int ct = 0; ct < 8; ++ct) {
        v16h vb;
        const h16* vrow = Vc + bch + (size_t)(ct * 16 + l15) * NPIX
                        + m0 + ms * 32 + (kh << 4);
        #pragma unroll
        for (int e = 0; e < 16; ++e) vb[e] = vrow[e];
        Oacc[ct] = wmma16(pa, vb, Oacc[ct]);
      }
    }
  }

  const float sc = scale_p[0];
  #pragma unroll
  for (int ct = 0; ct < 8; ++ct) {
    #pragma unroll
    for (int r = 0; r < 8; ++r) {
      int c = ct * 16 + l15;
      int n = n0 + wv * 16 + rbase + r;
      size_t idx = bch + (size_t)c * NPIX + n;
      float val = Oacc[ct][r] / lrow[r] * sc + Tadd[idx];
      Eout[idx] = val;
      E16[idx]  = (h16)val;
    }
  }
}

// ---------------------------------------------------------------------------
// 3x3 conv over cat([E_q, E_s]) as implicit GEMM (K = 256*9), fused BN + ReLU.
// ---------------------------------------------------------------------------
__global__ void __launch_bounds__(128)
conv3x3_bn_relu(const h16* __restrict__ Eq, const h16* __restrict__ Es,
                const float* __restrict__ Wc,
                const float* __restrict__ bg, const float* __restrict__ bb,
                const float* __restrict__ bm, const float* __restrict__ bv,
                float* __restrict__ outp)
{
  __shared__ h16 bt[64][40];              // im2col tile [pixel][k]
  const int tid  = threadIdx.x;
  const int lane = tid & 31;
  const int wv   = tid >> 5;
  const int kh   = lane >> 4;
  const int l15  = lane & 15;
  const int n0   = blockIdx.x * 64;
  const int o0   = blockIdx.y * 16;
  const int b    = blockIdx.z;
  const size_t bofs = (size_t)b * CI * NPIX;
  const float* Wrow = Wc + (size_t)(o0 + l15) * (CIN * 9);

  v8f acc = {};
  for (int k0 = 0; k0 < CIN * 9; k0 += 32) {
    __syncthreads();
    #pragma unroll
    for (int it = 0; it < 16; ++it) {
      int idx = tid + it * 128;
      int nl = idx & 63, kl = idx >> 6;
      int k  = k0 + kl;
      int ci = k / 9, rem = k - ci * 9;
      int ky = rem / 3 - 1, kx = rem - (rem / 3) * 3 - 1;
      int n = n0 + nl;
      int y = (n >> 6) + ky, x = (n & 63) + kx;
      h16 val = (h16)0.f;
      if (y >= 0 && y < 64 && x >= 0 && x < 64) {
        const h16* src = (ci < CI) ? Eq : Es;
        val = src[bofs + (size_t)(ci & (CI - 1)) * NPIX + y * 64 + x];
      }
      bt[nl][kl] = val;
    }
    __syncthreads();
    v16h a, bf;
    #pragma unroll
    for (int e = 0; e < 16; ++e) a[e]  = (h16)Wrow[k0 + a_kidx(e, kh)];
    #pragma unroll
    for (int e = 0; e < 16; ++e) bf[e] = bt[wv * 16 + l15][(kh << 4) + e];
    acc = wmma16(a, bf, acc);
  }

  const int rbase = kh * 8;
  #pragma unroll
  for (int r = 0; r < 8; ++r) {
    int o = o0 + rbase + r;
    int n = n0 + wv * 16 + l15;
    float inv = bg[o] * rsqrtf(bv[o] + BEPS);
    float x = (acc[r] - bm[o]) * inv + bb[o];
    outp[(size_t)b * CI * NPIX + (size_t)o * NPIX + n] = fmaxf(x, 0.f);
  }
}

// ---------------------------------------------------------------------------
extern "C" void kernel_launch(void* const* d_in, const int* in_sizes, int n_in,
                              void* d_out, int out_size, void* d_ws, size_t ws_size,
                              hipStream_t stream) {
  (void)in_sizes; (void)n_in; (void)out_size; (void)ws_size;
  const float* q     = (const float*)d_in[0];
  const float* s     = (const float*)d_in[1];
  const float* scale = (const float*)d_in[2];
  const float* w_v   = (const float*)d_in[3];  const float* b_v  = (const float*)d_in[4];
  const float* w_k1  = (const float*)d_in[5];  const float* b_k1 = (const float*)d_in[6];
  const float* w_q1  = (const float*)d_in[7];  const float* b_q1 = (const float*)d_in[8];
  const float* w_k2  = (const float*)d_in[9];  const float* b_k2 = (const float*)d_in[10];
  const float* w_q2  = (const float*)d_in[11]; const float* b_q2 = (const float*)d_in[12];
  const float* w_ts  = (const float*)d_in[13]; const float* b_ts = (const float*)d_in[14];
  const float* g_ts  = (const float*)d_in[15]; const float* be_ts= (const float*)d_in[16];
  const float* m_ts  = (const float*)d_in[17]; const float* v_ts = (const float*)d_in[18];
  const float* w_tq  = (const float*)d_in[19]; const float* b_tq = (const float*)d_in[20];
  const float* g_tq  = (const float*)d_in[21]; const float* be_tq= (const float*)d_in[22];
  const float* m_tq  = (const float*)d_in[23]; const float* v_tq = (const float*)d_in[24];
  const float* w_cat = (const float*)d_in[25];
  const float* g_cat = (const float*)d_in[26]; const float* be_cat=(const float*)d_in[27];
  const float* m_cat = (const float*)d_in[28]; const float* v_cat =(const float*)d_in[29];

  const int Bb = 2;
  char* ws = (char*)d_ws;
  size_t off = 0;
  auto take = [&](size_t bytes) { void* p = ws + off; off += (bytes + 255) & ~(size_t)255; return p; };
  h16*   vTq  = (h16*)  take((size_t)Bb * CI * NPIX * 2);
  h16*   vTs  = (h16*)  take((size_t)Bb * CI * NPIX * 2);
  h16*   kT   = (h16*)  take((size_t)Bb * NPIX * CI * 2);
  h16*   qT   = (h16*)  take((size_t)Bb * NPIX * CI * 2);
  float* st   = (float*)take((size_t)Bb * CI * NPIX * 4);
  float* qt   = (float*)take((size_t)Bb * CI * NPIX * 4);
  h16*   eq16 = (h16*)  take((size_t)Bb * CI * NPIX * 2);
  h16*   es16 = (h16*)  take((size_t)Bb * CI * NPIX * 2);

  float* outp   = (float*)d_out;
  float* Eq_out = outp + (size_t)Bb * CI * NPIX;
  float* Es_out = outp + 2 * (size_t)Bb * CI * NPIX;

  dim3 blk(128);
  dim3 g128(NPIX / 64, 8, Bb), g64(NPIX / 64, 4, Bb);

  // projections (layouts chosen for downstream WMMA fragment maps)
  proj_gemm<<<g128, blk, 0, stream>>>(q, w_v, b_v, nullptr, nullptr, nullptr, nullptr,
                                      nullptr, vTq, nullptr, CI, 0);
  proj_gemm<<<g128, blk, 0, stream>>>(s, w_v, b_v, nullptr, nullptr, nullptr, nullptr,
                                      nullptr, vTs, nullptr, CI, 0);
  proj_gemm<<<g64,  blk, 0, stream>>>(q, w_k1, b_k1, nullptr, nullptr, nullptr, nullptr,
                                      nullptr, nullptr, kT, 64, 0);
  proj_gemm<<<g64,  blk, 0, stream>>>(s, w_k2, b_k2, nullptr, nullptr, nullptr, nullptr,
                                      nullptr, nullptr, kT, 64, 64);
  proj_gemm<<<g64,  blk, 0, stream>>>(q, w_q1, b_q1, nullptr, nullptr, nullptr, nullptr,
                                      nullptr, nullptr, qT, 64, 0);
  proj_gemm<<<g64,  blk, 0, stream>>>(s, w_q2, b_q2, nullptr, nullptr, nullptr, nullptr,
                                      nullptr, nullptr, qT, 64, 64);
  proj_gemm<<<g128, blk, 0, stream>>>(s, w_ts, b_ts, g_ts, be_ts, m_ts, v_ts,
                                      st, nullptr, nullptr, CI, 0);
  proj_gemm<<<g128, blk, 0, stream>>>(q, w_tq, b_tq, g_tq, be_tq, m_tq, v_tq,
                                      qt, nullptr, nullptr, CI, 0);

  // fused attention both directions: E_s = sc*softmax(K·Qt)·Vs + s_t ; E_q symmetric
  dim3 gF(NPIX / 64, Bb);
  flash_attn<<<gF, blk, 0, stream>>>(kT, qT, vTs, st, scale, Es_out, es16);
  flash_attn<<<gF, blk, 0, stream>>>(qT, kT, vTq, qt, scale, Eq_out, eq16);

  // 3x3 conv + BN + ReLU -> cpam
  dim3 gC(NPIX / 64, 8, Bb);
  conv3x3_bn_relu<<<gC, blk, 0, stream>>>(eq16, es16, w_cat,
                                          g_cat, be_cat, m_cat, v_cat, outp);
}
